// YOLOspineV2_10230612099539
// MI455X (gfx1250) — matrix operand order repaired and checked by reading, not compiled, not run on hardware
//
#include <hip/hip_runtime.h>
#include <stdint.h>
#include <vector>

// ---------------------------------------------------------------------------
// Types
// ---------------------------------------------------------------------------
typedef __bf16 bf16_t;
typedef __attribute__((ext_vector_type(16))) __bf16 v16bf;
typedef __attribute__((ext_vector_type(8)))  float  v8f;
typedef unsigned int uint32x4 __attribute__((ext_vector_type(4)));
typedef int          int32x4  __attribute__((ext_vector_type(4)));
typedef int          int32x8  __attribute__((ext_vector_type(8)));

#define BM 64
#define BN 128
#define BK 32
static const int NB   = 2;     // batch
static const int IMGS = 384;
static const int KTOP = 100;

// ---------------------------------------------------------------------------
// TDM helper: 2D tile (2-byte elements) global -> LDS.  D# per CDNA5 ISA ch.8.
// tensor dims = remaining extent so OOB reads zero-fill the tile edges.
// 6-arg builtin (clang-23 / therock-10.0 toolchain).
// ---------------------------------------------------------------------------
__device__ inline void tdm_load_2d_b16(uint32_t lds_off, const void* gaddr,
                                       uint32_t tens_w, uint32_t tens_h,
                                       uint32_t tile_w, uint32_t tile_h,
                                       uint64_t row_stride_elems) {
  uint64_t ga = (uint64_t)(uintptr_t)gaddr;
  uint32x4 g0 = { 1u,                                   // count=1, user mode
                  lds_off,                              // lds_addr
                  (uint32_t)(ga & 0xFFFFFFFFu),
                  (uint32_t)((ga >> 32) & 0x1FFFFFFu) | (2u << 30) }; // type=2
  int32x8 g1 = {
    (int)(1u << 16),                                            // data_size=1 (2B)
    (int)((tens_w & 0xFFFFu) << 16),                            // tensor_dim0 lo
    (int)(((tens_w >> 16) & 0xFFFFu) | ((tens_h & 0xFFFFu) << 16)),
    (int)(((tens_h >> 16) & 0xFFFFu) | ((tile_w & 0xFFFFu) << 16)),
    (int)(tile_h & 0xFFFFu),                                    // tile_dim1 (dim2=0)
    (int)(uint32_t)(row_stride_elems & 0xFFFFFFFFu),            // dim0 stride lo
    (int)(uint32_t)((row_stride_elems >> 32) & 0xFFFFu),        // dim0 stride hi
    0 };
  int32x4 z4 = {0, 0, 0, 0};
  int32x8 z8 = {0, 0, 0, 0, 0, 0, 0, 0};
  __builtin_amdgcn_tensor_load_to_lds(g0, g1, z4, z4, z8, 0);
}

__device__ inline float sigmoidf_(float v) { return 1.f / (1.f + __expf(-v)); }

// ---------------------------------------------------------------------------
// Elementwise / utility kernels
// ---------------------------------------------------------------------------
__global__ void k_cast_bf16(const float* __restrict__ s, bf16_t* __restrict__ d, size_t n) {
  size_t i = (size_t)blockIdx.x * 256 + threadIdx.x;
  if (i < n) d[i] = (bf16_t)s[i];
}

// w[Cout][K] fp32  ->  o[K][Cout] bf16
__global__ void k_repack_w(const float* __restrict__ w, bf16_t* __restrict__ o,
                           int Cout, int K) {
  size_t i = (size_t)blockIdx.x * 256 + threadIdx.x;
  if (i >= (size_t)Cout * K) return;
  int k  = (int)(i % K);
  int co = (int)(i / K);
  o[(size_t)k * Cout + co] = (bf16_t)w[i];
}

// per-channel batch stats over (N, H, W)
__global__ void k_stats(const float* __restrict__ y, float* __restrict__ mean,
                        float* __restrict__ rstd, int N, int C, int HW, float eps) {
  int c = blockIdx.x;
  float s = 0.f, s2 = 0.f;
  for (int n = 0; n < N; ++n) {
    const float* p = y + ((size_t)n * C + c) * HW;
    for (int i = threadIdx.x; i < HW; i += 256) { float v = p[i]; s += v; s2 += v * v; }
  }
  __shared__ float sh[256], sh2[256];
  sh[threadIdx.x] = s; sh2[threadIdx.x] = s2;
  __syncthreads();
  for (int o = 128; o > 0; o >>= 1) {
    if (threadIdx.x < (unsigned)o) { sh[threadIdx.x] += sh[threadIdx.x + o]; sh2[threadIdx.x] += sh2[threadIdx.x + o]; }
    __syncthreads();
  }
  if (threadIdx.x == 0) {
    float cnt = (float)N * (float)HW;
    float m = sh[0] / cnt;
    float v = sh2[0] / cnt - m * m;
    mean[c] = m;
    rstd[c] = rsqrtf(v + eps);
  }
}

// y (contig f32) -> SiLU(BN(y)) bf16 into strided channel slice
__global__ void k_bn_silu(const float* __restrict__ y, const float* __restrict__ mean,
                          const float* __restrict__ rstd, const float* __restrict__ g,
                          const float* __restrict__ b, bf16_t* __restrict__ dst,
                          int dc0, int dCt, int N, int C, int HW) {
  size_t i = (size_t)blockIdx.x * 256 + threadIdx.x;
  size_t tot = (size_t)N * C * HW;
  if (i >= tot) return;
  int hw = (int)(i % HW);
  size_t t = i / HW;
  int c = (int)(t % C);
  int n = (int)(t / C);
  float v = y[i];
  v = (v - mean[c]) * rstd[c] * g[c] + b[c];
  v = v * sigmoidf_(v);
  dst[(((size_t)n * dCt + dc0 + c) * HW) + hw] = (bf16_t)v;
}

// dst = a(strided) + b(contig), strided dst
__global__ void k_add(const bf16_t* __restrict__ a, int ac0, int aCt,
                      const bf16_t* __restrict__ bb, bf16_t* __restrict__ dst,
                      int dc0, int dCt, int N, int C, int HW) {
  size_t i = (size_t)blockIdx.x * 256 + threadIdx.x;
  size_t tot = (size_t)N * C * HW;
  if (i >= tot) return;
  int hw = (int)(i % HW);
  size_t t = i / HW;
  int c = (int)(t % C);
  int n = (int)(t / C);
  float va = (float)a[(((size_t)n * aCt + ac0 + c) * HW) + hw];
  float vb = (float)bb[(((size_t)n * C + c) * HW) + hw];
  dst[(((size_t)n * dCt + dc0 + c) * HW) + hw] = (bf16_t)(va + vb);
}

__global__ void k_copy_slice(const bf16_t* __restrict__ src, int sc0, int sCt,
                             bf16_t* __restrict__ dst, int dc0, int dCt,
                             int N, int C, int HW) {
  size_t i = (size_t)blockIdx.x * 256 + threadIdx.x;
  size_t tot = (size_t)N * C * HW;
  if (i >= tot) return;
  int hw = (int)(i % HW);
  size_t t = i / HW;
  int c = (int)(t % C);
  int n = (int)(t / C);
  dst[(((size_t)n * dCt + dc0 + c) * HW) + hw] =
      src[(((size_t)n * sCt + sc0 + c) * HW) + hw];
}

__global__ void k_upsample2(const bf16_t* __restrict__ src, int sc0, int sCt,
                            bf16_t* __restrict__ dst, int dc0, int dCt,
                            int N, int C, int H, int W) {
  int H2 = 2 * H, W2 = 2 * W;
  size_t i = (size_t)blockIdx.x * 256 + threadIdx.x;
  size_t tot = (size_t)N * C * H2 * W2;
  if (i >= tot) return;
  int x = (int)(i % W2);
  size_t t = i / W2;
  int y = (int)(t % H2); t /= H2;
  int c = (int)(t % C);
  int n = (int)(t / C);
  dst[(((size_t)n * dCt + dc0 + c) * H2 + y) * W2 + x] =
      src[(((size_t)n * sCt + sc0 + c) * H + (y >> 1)) * W + (x >> 1)];
}

__global__ void k_maxpool5(const bf16_t* __restrict__ src, int sc0, int sCt,
                           bf16_t* __restrict__ dst, int dc0, int dCt,
                           int N, int C, int H, int W) {
  size_t i = (size_t)blockIdx.x * 256 + threadIdx.x;
  size_t tot = (size_t)N * C * H * W;
  if (i >= tot) return;
  int x = (int)(i % W);
  size_t t = i / W;
  int y = (int)(t % H); t /= H;
  int c = (int)(t % C);
  int n = (int)(t / C);
  const bf16_t* p = src + ((size_t)n * sCt + sc0 + c) * H * W;
  float m = -1e30f;
  for (int dy = -2; dy <= 2; ++dy) {
    int yy = y + dy;
    if (yy < 0 || yy >= H) continue;
    for (int dx = -2; dx <= 2; ++dx) {
      int xx = x + dx;
      if (xx < 0 || xx >= W) continue;
      float v = (float)p[yy * W + xx];
      m = fmaxf(m, v);
    }
  }
  dst[(((size_t)n * dCt + dc0 + c) * H + y) * W + x] = (bf16_t)m;
}

// small 1x1 conv with bias (det output heads), fp32 out into strided channels
__global__ void k_lin1x1(const bf16_t* __restrict__ x, const float* __restrict__ w,
                         const float* __restrict__ bias, float* __restrict__ out,
                         int oc0, int oCt, int N, int Cin, int Cout, int HW) {
  size_t i = (size_t)blockIdx.x * 256 + threadIdx.x;
  size_t tot = (size_t)N * Cout * HW;
  if (i >= tot) return;
  int hw = (int)(i % HW);
  size_t t = i / HW;
  int c = (int)(t % Cout);
  int n = (int)(t / Cout);
  float s = bias[c];
  const float* wr = w + (size_t)c * Cin;
  const bf16_t* xp = x + (size_t)n * Cin * HW + hw;
  for (int ci = 0; ci < Cin; ++ci) s += (float)xp[(size_t)ci * HW] * wr[ci];
  out[(((size_t)n * oCt + oc0 + c) * HW) + hw] = s;
}

// ---------------------------------------------------------------------------
// Shared WMMA core: each wave owns a 16(M) x 64(N) strip => 4 accumulators,
// one A fragment reused across 4 back-to-back WMMAs per 32-K step.
// Block tile 64x128, 8 waves as 4(M) x 2(N).
// ---------------------------------------------------------------------------
struct WaveFrag {
  v16bf a;
  v16bf b0, b1, b2, b3;
};

__device__ inline void load_frags(const bf16_t lsA[BM][BK + 4],
                                  const bf16_t lsB[BK][BN],
                                  int wm, int wn, int lane, WaveFrag& f) {
  const int ml = wm * 16 + (lane & 15);
  const int kb = (lane >> 4) * 8;
#pragma unroll
  for (int v = 0; v < 8; ++v) {
    int k = ((v & 4) ? 16 : 0) + kb + (v & 3) * 2;
    f.a[2 * v]     = lsA[ml][k];
    f.a[2 * v + 1] = lsA[ml][k + 1];
  }
  const int nbase = wn * 64 + (lane & 15);
  const int kb2 = (lane >> 4) * 8;
#pragma unroll
  for (int v = 0; v < 8; ++v) {
    int k0 = ((v & 4) ? 16 : 0) + kb2 + (v & 3) * 2, k1 = k0 + 1;
    f.b0[2 * v]     = lsB[k0][nbase];
    f.b0[2 * v + 1] = lsB[k1][nbase];
    f.b1[2 * v]     = lsB[k0][nbase + 16];
    f.b1[2 * v + 1] = lsB[k1][nbase + 16];
    f.b2[2 * v]     = lsB[k0][nbase + 32];
    f.b2[2 * v + 1] = lsB[k1][nbase + 32];
    f.b3[2 * v]     = lsB[k0][nbase + 48];
    f.b3[2 * v + 1] = lsB[k1][nbase + 48];
  }
}

// ---------------------------------------------------------------------------
// WMMA implicit-GEMM convolution (bf16 -> f32)
//   A[m,k] = im2col(x), k = (ci*KH + ky)*KW + kx ; B = wt[K][Cout]
// ---------------------------------------------------------------------------
__global__ __launch_bounds__(256) void k_conv_wmma(
    const bf16_t* __restrict__ x, int xc0, int xCt,
    const bf16_t* __restrict__ wt, float* __restrict__ y,
    int N, int Cin, int H, int W,
    int Cout, int KH, int KW, int stride, int pad, int OH, int OW) {
  __shared__ bf16_t lsA[BM][BK + 4];
  __shared__ bf16_t lsB[BK][BN];
  const int M = N * OH * OW;
  const int Ktot = Cin * KH * KW;
  const int m0 = blockIdx.x * BM;
  const int n0 = blockIdx.y * BN;
  const int t = threadIdx.x;
  const int lane = t & 31;
  const int wv = t >> 5;
  const int wm = wv >> 1, wn = wv & 1;

  v8f acc0 = {}, acc1 = {}, acc2 = {}, acc3 = {};

  const int ksteps = (Ktot + BK - 1) / BK;
  for (int kt = 0; kt < ksteps; ++kt) {
    const int k0 = kt * BK;
    __syncthreads();
    if (wv == 0) {
      // async B tile via Tensor Data Mover (zero-fills OOB edges)
      tdm_load_2d_b16((uint32_t)(uintptr_t)(&lsB[0][0]),
                      wt + (size_t)k0 * Cout + n0,
                      (uint32_t)(Cout - n0), (uint32_t)(Ktot - k0),
                      (uint32_t)BN, (uint32_t)BK, (uint64_t)Cout);
    }
    // prefetch next K-tile of weights into GL2
    __builtin_prefetch((const void*)(wt + (size_t)(k0 + BK) * Cout), 0, 1);
    // A tile gather (im2col, zero-padded)
    for (int e = t; e < BM * BK; e += 256) {
      int row = e / BK, col = e % BK;
      int gm = m0 + row, gk = k0 + col;
      bf16_t vv = (bf16_t)0.0f;
      if (gm < M && gk < Ktot) {
        int ow_ = gm % OW; int tm = gm / OW;
        int oh_ = tm % OH; int nb = tm / OH;
        int kw_ = gk % KW; int tk = gk / KW;
        int kh_ = tk % KH; int ci = tk / KH;
        int iy = oh_ * stride + kh_ - pad;
        int ix = ow_ * stride + kw_ - pad;
        if (iy >= 0 && iy < H && ix >= 0 && ix < W)
          vv = x[(((size_t)nb * xCt + xc0 + ci) * H + iy) * W + ix];
      }
      lsA[row][col] = vv;
    }
    if (wv == 0) __builtin_amdgcn_s_wait_tensorcnt(0);
    __syncthreads();

    WaveFrag f;
    load_frags(lsA, lsB, wm, wn, lane, f);
    acc0 = __builtin_amdgcn_wmma_f32_16x16x32_bf16(false, f.a, false, f.b0, (short)0, acc0, false, false);
    acc1 = __builtin_amdgcn_wmma_f32_16x16x32_bf16(false, f.a, false, f.b1, (short)0, acc1, false, false);
    acc2 = __builtin_amdgcn_wmma_f32_16x16x32_bf16(false, f.a, false, f.b2, (short)0, acc2, false, false);
    acc3 = __builtin_amdgcn_wmma_f32_16x16x32_bf16(false, f.a, false, f.b3, (short)0, acc3, false, false);
  }

  const int nl = lane & 15;
#pragma unroll
  for (int j = 0; j < 4; ++j) {
    v8f acc = (j == 0) ? acc0 : (j == 1) ? acc1 : (j == 2) ? acc2 : acc3;
    int oc = n0 + wn * 64 + j * 16 + nl;
    if (oc >= Cout) continue;
#pragma unroll
    for (int v = 0; v < 8; ++v) {
      int m = wm * 16 + v + 8 * (lane >> 4);
      int gm = m0 + m;
      if (gm < M) {
        int ow_ = gm % OW; int tm = gm / OW;
        int oh_ = tm % OH; int nb = tm / OH;
        y[(((size_t)nb * Cout + oc) * OH + oh_) * OW + ow_] = acc[v];
      }
    }
  }
}

// ---------------------------------------------------------------------------
// WMMA GEMM: C[M,Nn] = A[M,K] * Bt[K,Nn] + bias  (A,Bt bf16, C f32)
// ---------------------------------------------------------------------------
__global__ __launch_bounds__(256) void k_gemm_wmma(
    const bf16_t* __restrict__ A, const bf16_t* __restrict__ Bt,
    const float* __restrict__ bias, float* __restrict__ Cm,
    int M, int K, int Nn) {
  __shared__ bf16_t lsA[BM][BK + 4];
  __shared__ bf16_t lsB[BK][BN];
  const int m0 = blockIdx.x * BM;
  const int n0 = blockIdx.y * BN;
  const int t = threadIdx.x;
  const int lane = t & 31;
  const int wv = t >> 5;
  const int wm = wv >> 1, wn = wv & 1;

  v8f acc0 = {}, acc1 = {}, acc2 = {}, acc3 = {};
  const int ksteps = (K + BK - 1) / BK;
  for (int kt = 0; kt < ksteps; ++kt) {
    const int k0 = kt * BK;
    __syncthreads();
    if (wv == 0) {
      tdm_load_2d_b16((uint32_t)(uintptr_t)(&lsB[0][0]),
                      Bt + (size_t)k0 * Nn + n0,
                      (uint32_t)(Nn - n0), (uint32_t)(K - k0),
                      (uint32_t)BN, (uint32_t)BK, (uint64_t)Nn);
    }
    for (int e = t; e < BM * BK; e += 256) {
      int row = e / BK, col = e % BK;
      int gm = m0 + row, gk = k0 + col;
      bf16_t vv = (bf16_t)0.0f;
      if (gm < M && gk < K) vv = A[(size_t)gm * K + gk];
      lsA[row][col] = vv;
    }
    if (wv == 0) __builtin_amdgcn_s_wait_tensorcnt(0);
    __syncthreads();

    WaveFrag f;
    load_frags(lsA, lsB, wm, wn, lane, f);
    acc0 = __builtin_amdgcn_wmma_f32_16x16x32_bf16(false, f.a, false, f.b0, (short)0, acc0, false, false);
    acc1 = __builtin_amdgcn_wmma_f32_16x16x32_bf16(false, f.a, false, f.b1, (short)0, acc1, false, false);
    acc2 = __builtin_amdgcn_wmma_f32_16x16x32_bf16(false, f.a, false, f.b2, (short)0, acc2, false, false);
    acc3 = __builtin_amdgcn_wmma_f32_16x16x32_bf16(false, f.a, false, f.b3, (short)0, acc3, false, false);
  }

  const int nl = lane & 15;
#pragma unroll
  for (int j = 0; j < 4; ++j) {
    v8f acc = (j == 0) ? acc0 : (j == 1) ? acc1 : (j == 2) ? acc2 : acc3;
    int oc = n0 + wn * 64 + j * 16 + nl;
    if (oc >= Nn) continue;
    float bv = bias ? bias[oc] : 0.f;
#pragma unroll
    for (int v = 0; v < 8; ++v) {
      int m = wm * 16 + v + 8 * (lane >> 4);
      int gm = m0 + m;
      if (gm < M) Cm[(size_t)gm * Nn + oc] = acc[v] + bv;
    }
  }
}

// ---------------------------------------------------------------------------
// top-100 per image per level + box decode + proposals
// ---------------------------------------------------------------------------
__global__ void k_extract(const float* __restrict__ det, int H, int W, float stride,
                          float* __restrict__ boxes, float* __restrict__ props) {
  const int b = blockIdx.x;
  const int HW = H * W;
  __shared__ float sc[2304];
  __shared__ float rv[256];
  __shared__ int   ri[256];
  const float* base = det + (size_t)b * 11 * HW;
  for (int i = threadIdx.x; i < HW; i += 256) {
    float mc = -1e30f;
    for (int c = 0; c < 6; ++c) mc = fmaxf(mc, sigmoidf_(base[c * HW + i]));
    sc[i] = mc * sigmoidf_(base[10 * HW + i]);
  }
  __syncthreads();
  for (int it = 0; it < KTOP; ++it) {
    float bv = -1e30f; int bi = 0;
    for (int i = threadIdx.x; i < HW; i += 256) {
      float v = sc[i];
      if (v > bv) { bv = v; bi = i; }
    }
    rv[threadIdx.x] = bv; ri[threadIdx.x] = bi;
    __syncthreads();
    for (int o = 128; o > 0; o >>= 1) {
      if (threadIdx.x < (unsigned)o) {
        if (rv[threadIdx.x + o] > rv[threadIdx.x] ||
            (rv[threadIdx.x + o] == rv[threadIdx.x] && ri[threadIdx.x + o] < ri[threadIdx.x])) {
          rv[threadIdx.x] = rv[threadIdx.x + o];
          ri[threadIdx.x] = ri[threadIdx.x + o];
        }
      }
      __syncthreads();
    }
    if (threadIdx.x == 0) {
      int idx = ri[0];
      sc[idx] = -1e30f;
      int gx = idx % W, gy = idx / W;
      float b0 = base[6 * HW + idx], b1 = base[7 * HW + idx];
      float b2 = base[8 * HW + idx], b3 = base[9 * HW + idx];
      float cx = (gx + 0.5f + b0) * stride;
      float cy = (gy + 0.5f + b1) * stride;
      float bw = __expf(fminf(fmaxf(b2, -8.f), 4.f)) * stride;
      float bh = __expf(fminf(fmaxf(b3, -8.f), 4.f)) * stride;
      float x1 = fminf(fmaxf(cx - bw * 0.5f, 0.f), (float)IMGS);
      float y1 = fminf(fmaxf(cy - bh * 0.5f, 0.f), (float)IMGS);
      float x2 = fminf(fmaxf(cx + bw * 0.5f, 0.f), (float)IMGS);
      float y2 = fminf(fmaxf(cy + bh * 0.5f, 0.f), (float)IMGS);
      size_t r = (size_t)b * KTOP + it;
      boxes[r * 4 + 0] = x1; boxes[r * 4 + 1] = y1;
      boxes[r * 4 + 2] = x2; boxes[r * 4 + 3] = y2;
      props[r * 5 + 0] = (float)b;
      props[r * 5 + 1] = x1; props[r * 5 + 2] = y1;
      props[r * 5 + 3] = x2; props[r * 5 + 4] = y2;
    }
    __syncthreads();
  }
}

// ---------------------------------------------------------------------------
// ROI align (7x7, SR=2) -> bf16 feature rows [R, C*49]
// ---------------------------------------------------------------------------
__global__ void k_roialign(const bf16_t* __restrict__ feat, const float* __restrict__ boxes,
                           int C, int H, int W, float scale,
                           bf16_t* __restrict__ out, int fin) {
  int r = blockIdx.x;
  int b = r / KTOP;
  float x1 = boxes[r * 4 + 0] * scale - 0.5f;
  float y1 = boxes[r * 4 + 1] * scale - 0.5f;
  float x2 = boxes[r * 4 + 2] * scale - 0.5f;
  float y2 = boxes[r * 4 + 3] * scale - 0.5f;
  float bw = (x2 - x1) / 7.f, bh = (y2 - y1) / 7.f;
  const bf16_t* f = feat + (size_t)b * C * H * W;
  for (int idx = threadIdx.x; idx < fin; idx += 256) {
    int c = idx / 49;
    int pp = idx % 49;
    int py = pp / 7, px = pp % 7;
    const bf16_t* fc = f + (size_t)c * H * W;
    float acc = 0.f;
#pragma unroll
    for (int sy = 0; sy < 2; ++sy)
#pragma unroll
      for (int sx = 0; sx < 2; ++sx) {
        float yy = y1 + ((float)py + ((float)sy + 0.5f) * 0.5f) * bh;
        float xx = x1 + ((float)px + ((float)sx + 0.5f) * 0.5f) * bw;
        bool valid = (yy > -1.f) && (yy < (float)H) && (xx > -1.f) && (xx < (float)W);
        float yc = fminf(fmaxf(yy, 0.f), (float)H - 1.f);
        float xc = fminf(fmaxf(xx, 0.f), (float)W - 1.f);
        int y0 = (int)floorf(yc), x0 = (int)floorf(xc);
        int yh = y0 + 1; if (yh > H - 1) yh = H - 1;
        int xh = x0 + 1; if (xh > W - 1) xh = W - 1;
        float ly = yc - (float)y0, lx = xc - (float)x0;
        float v = (float)fc[y0 * W + x0] * (1.f - ly) * (1.f - lx)
                + (float)fc[y0 * W + xh] * (1.f - ly) * lx
                + (float)fc[yh * W + x0] * ly * (1.f - lx)
                + (float)fc[yh * W + xh] * ly * lx;
        acc += valid ? v : 0.f;
      }
    out[(size_t)r * fin + idx] = (bf16_t)(acc * 0.25f);
  }
}

// layernorm over last dim + relu -> bf16
__global__ void k_ln_relu(const float* __restrict__ in, const float* __restrict__ g,
                          const float* __restrict__ b, bf16_t* __restrict__ out, int Nn) {
  int m = blockIdx.x;
  const float* row = in + (size_t)m * Nn;
  __shared__ float sh[256], sh2[256];
  float s = 0.f, s2 = 0.f;
  for (int i = threadIdx.x; i < Nn; i += 256) { float v = row[i]; s += v; s2 += v * v; }
  sh[threadIdx.x] = s; sh2[threadIdx.x] = s2;
  __syncthreads();
  for (int o = 128; o > 0; o >>= 1) {
    if (threadIdx.x < (unsigned)o) { sh[threadIdx.x] += sh[threadIdx.x + o]; sh2[threadIdx.x] += sh2[threadIdx.x + o]; }
    __syncthreads();
  }
  __shared__ float mu, rs;
  if (threadIdx.x == 0) {
    mu = sh[0] / (float)Nn;
    float var = sh2[0] / (float)Nn - mu * mu;
    rs = rsqrtf(var + 1e-5f);
  }
  __syncthreads();
  for (int i = threadIdx.x; i < Nn; i += 256) {
    float v = (row[i] - mu) * rs * g[i] + b[i];
    out[(size_t)m * Nn + i] = (bf16_t)fmaxf(v, 0.f);
  }
}

// final refine heads: out[m, 0:6]=h@wc^T+bc ; out[m, 6:10]=h@wr^T+br
__global__ void k_linear10(const bf16_t* __restrict__ h, const float* __restrict__ wc,
                           const float* __restrict__ bc, const float* __restrict__ wr,
                           const float* __restrict__ br, float* __restrict__ out, int M) {
  int i = blockIdx.x * 256 + threadIdx.x;
  if (i >= M * 10) return;
  int m = i / 10, j = i % 10;
  const float* wrow;
  float bb;
  if (j < 6) { wrow = wc + (size_t)j * 256; bb = bc[j]; }
  else       { wrow = wr + (size_t)(j - 6) * 256; bb = br[j - 6]; }
  const bf16_t* hr = h + (size_t)m * 256;
  float s = bb;
  for (int q = 0; q < 256; ++q) s += (float)hr[q] * wrow[q];
  out[(size_t)m * 10 + j] = s;
}

// ---------------------------------------------------------------------------
// Host-side orchestration
// ---------------------------------------------------------------------------
namespace {

struct Runner {
  hipStream_t s;
  char* ws;
  size_t cap, off;
  void* alloc(size_t bytes) {
    bytes = (bytes + 255) & ~(size_t)255;
    if (off + bytes > cap) off = 0;   // ring fallback
    void* p = ws + off;
    off += bytes;
    return p;
  }
  float*  af(size_t n) { return (float*)alloc(n * 4); }
  bf16_t* ah(size_t n) { return (bf16_t*)alloc(n * 2); }
};

struct Cur {
  void* const* din;
  int i;
  const float* nf() { return (const float*)din[i++]; }
};

struct CBS { const float *w, *g, *b; };
static CBS takeCBS(Cur& c) { CBS p; p.w = c.nf(); p.g = c.nf(); p.b = c.nf(); return p; }

struct Act { bf16_t* p; int c0, Ct, C, H, W; };

static inline unsigned gr1(size_t n) { return (unsigned)((n + 255) / 256); }

static Act cbs_apply(Runner& R, const CBS& p, Act in, int Cout, int k, int stride,
                     bf16_t* dst, int dc0, int dCt) {
  int K = in.C * k * k;
  bf16_t* wt = R.ah((size_t)K * Cout);
  k_repack_w<<<gr1((size_t)K * Cout), 256, 0, R.s>>>(p.w, wt, Cout, K);
  int OH = in.H / stride, OW = in.W / stride;
  float* tmp = R.af((size_t)NB * Cout * OH * OW);
  {
    int M = NB * OH * OW;
    dim3 grid((M + BM - 1) / BM, (Cout + BN - 1) / BN);
    k_conv_wmma<<<grid, 256, 0, R.s>>>(in.p, in.c0, in.Ct, wt, tmp,
                                       NB, in.C, in.H, in.W,
                                       Cout, k, k, stride, k / 2, OH, OW);
  }
  float* mean = R.af(Cout);
  float* rstd = R.af(Cout);
  k_stats<<<Cout, 256, 0, R.s>>>(tmp, mean, rstd, NB, Cout, OH * OW, 1e-3f);
  if (!dst) { dst = R.ah((size_t)NB * Cout * OH * OW); dc0 = 0; dCt = Cout; }
  size_t tot = (size_t)NB * Cout * OH * OW;
  k_bn_silu<<<gr1(tot), 256, 0, R.s>>>(tmp, mean, rstd, p.g, p.b, dst, dc0, dCt,
                                       NB, Cout, OH * OW);
  Act o; o.p = dst; o.c0 = dc0; o.Ct = dCt; o.C = Cout; o.H = OH; o.W = OW;
  return o;
}

static Act c2f_run(Runner& R, Cur& cur, Act in, int Cout, int nblk, bool shortcut) {
  int c = Cout / 2, H = in.H, W = in.W, HW = H * W;
  CBS cv1 = takeCBS(cur), cv2 = takeCBS(cur);
  std::vector<CBS> m1(nblk), m2(nblk);
  for (int i = 0; i < nblk; ++i) { m1[i] = takeCBS(cur); m2[i] = takeCBS(cur); }
  int Ct = (2 + nblk) * c;
  bf16_t* cat = R.ah((size_t)NB * Ct * HW);
  cbs_apply(R, cv1, in, 2 * c, 1, 1, cat, 0, Ct);
  for (int i = 0; i < nblk; ++i) {
    Act z; z.p = cat; z.c0 = (1 + i) * c; z.Ct = Ct; z.C = c; z.H = H; z.W = W;
    Act tA = cbs_apply(R, m1[i], z, c / 2, 1, 1, nullptr, 0, 0);
    if (shortcut) {
      Act o = cbs_apply(R, m2[i], tA, c, 3, 1, nullptr, 0, 0);
      size_t tot = (size_t)NB * c * HW;
      k_add<<<gr1(tot), 256, 0, R.s>>>(cat, (1 + i) * c, Ct, o.p, cat,
                                       (2 + i) * c, Ct, NB, c, HW);
    } else {
      cbs_apply(R, m2[i], tA, c, 3, 1, cat, (2 + i) * c, Ct);
    }
  }
  Act catA; catA.p = cat; catA.c0 = 0; catA.Ct = Ct; catA.C = Ct; catA.H = H; catA.W = W;
  return cbs_apply(R, cv2, catA, Cout, 1, 1, nullptr, 0, 0);
}

static Act sppf_run(Runner& R, Cur& cur, Act in, int Cout) {
  CBS cv1 = takeCBS(cur), cv2 = takeCBS(cur);
  int h = in.C / 2, H = in.H, W = in.W, HW = H * W, Ct = 4 * h;
  bf16_t* cat = R.ah((size_t)NB * Ct * HW);
  cbs_apply(R, cv1, in, h, 1, 1, cat, 0, Ct);
  size_t tot = (size_t)NB * h * HW;
  for (int j = 0; j < 3; ++j)
    k_maxpool5<<<gr1(tot), 256, 0, R.s>>>(cat, j * h, Ct, cat, (j + 1) * h, Ct,
                                          NB, h, H, W);
  Act catA; catA.p = cat; catA.c0 = 0; catA.Ct = Ct; catA.C = Ct; catA.H = H; catA.W = W;
  return cbs_apply(R, cv2, catA, Cout, 1, 1, nullptr, 0, 0);
}

static void run_det(Runner& R, Cur& cur, Act f, int c, float* detOut) {
  int HW = f.H * f.W;
  Act t1 = cbs_apply(R, takeCBS(cur), f, c, 3, 1, nullptr, 0, 0);
  Act t2 = cbs_apply(R, takeCBS(cur), t1, c, 3, 1, nullptr, 0, 0);
  // cls
  Act ac = cbs_apply(R, takeCBS(cur), t2, c, 3, 1, nullptr, 0, 0);
  const float* clsw = cur.nf(); const float* clsb = cur.nf();
  k_lin1x1<<<gr1((size_t)NB * 6 * HW), 256, 0, R.s>>>(ac.p, clsw, clsb, detOut,
                                                      0, 11, NB, c, 6, HW);
  // reg
  Act ar = cbs_apply(R, takeCBS(cur), t2, c, 3, 1, nullptr, 0, 0);
  const float* regw = cur.nf(); const float* regb = cur.nf();
  k_lin1x1<<<gr1((size_t)NB * 4 * HW), 256, 0, R.s>>>(ar.p, regw, regb, detOut,
                                                      6, 11, NB, c, 4, HW);
  // obj
  Act ao = cbs_apply(R, takeCBS(cur), t2, c / 2, 3, 1, nullptr, 0, 0);
  const float* objw = cur.nf(); const float* objb = cur.nf();
  k_lin1x1<<<gr1((size_t)NB * 1 * HW), 256, 0, R.s>>>(ao.p, objw, objb, detOut,
                                                      10, 11, NB, c / 2, 1, HW);
}

static void run_post(Runner& R, Cur& cur, const float* det, Act feat, float stride,
                     float* refOut, float* propOut) {
  int H = feat.H, W = feat.W;
  float* boxes = R.af((size_t)NB * KTOP * 4);
  k_extract<<<NB, 256, 0, R.s>>>(det, H, W, stride, boxes, propOut);
  int fin = feat.C * 49;
  int Mr = NB * KTOP;
  bf16_t* roi = R.ah((size_t)Mr * fin);
  k_roialign<<<Mr, 256, 0, R.s>>>(feat.p, boxes, feat.C, H, W, 1.f / stride, roi, fin);

  const float* w1 = cur.nf(); const float* b1 = cur.nf();
  const float* g1 = cur.nf(); const float* bb1 = cur.nf();
  const float* w2 = cur.nf(); const float* b2 = cur.nf();
  const float* g2 = cur.nf(); const float* bb2 = cur.nf();
  const float* wc = cur.nf(); const float* bc = cur.nf();
  const float* wr = cur.nf(); const float* br = cur.nf();

  dim3 grid((Mr + BM - 1) / BM, (256 + BN - 1) / BN);
  // layer 1
  bf16_t* w1t = R.ah((size_t)fin * 256);
  k_repack_w<<<gr1((size_t)fin * 256), 256, 0, R.s>>>(w1, w1t, 256, fin);
  float* h1 = R.af((size_t)Mr * 256);
  k_gemm_wmma<<<grid, 256, 0, R.s>>>(roi, w1t, b1, h1, Mr, fin, 256);
  bf16_t* h1h = R.ah((size_t)Mr * 256);
  k_ln_relu<<<Mr, 256, 0, R.s>>>(h1, g1, bb1, h1h, 256);
  // layer 2
  bf16_t* w2t = R.ah((size_t)256 * 256);
  k_repack_w<<<gr1((size_t)256 * 256), 256, 0, R.s>>>(w2, w2t, 256, 256);
  float* h2 = R.af((size_t)Mr * 256);
  k_gemm_wmma<<<grid, 256, 0, R.s>>>(h1h, w2t, b2, h2, Mr, 256, 256);
  bf16_t* h2h = R.ah((size_t)Mr * 256);
  k_ln_relu<<<Mr, 256, 0, R.s>>>(h2, g2, bb2, h2h, 256);
  // heads
  k_linear10<<<gr1((size_t)Mr * 10), 256, 0, R.s>>>(h2h, wc, bc, wr, br, refOut, Mr);
}

}  // namespace

extern "C" void kernel_launch(void* const* d_in, const int* in_sizes, int n_in,
                              void* d_out, int out_size, void* d_ws, size_t ws_size,
                              hipStream_t stream) {
  (void)in_sizes; (void)n_in; (void)out_size;
  Runner R; R.s = stream; R.ws = (char*)d_ws; R.cap = ws_size; R.off = 0;
  Cur cur; cur.din = d_in; cur.i = 1;

  // input cast
  const float* xin = (const float*)d_in[0];
  size_t xn = (size_t)NB * 3 * IMGS * IMGS;
  bf16_t* xh = R.ah(xn);
  k_cast_bf16<<<gr1(xn), 256, 0, stream>>>(xin, xh, xn);
  Act x0; x0.p = xh; x0.c0 = 0; x0.Ct = 3; x0.C = 3; x0.H = IMGS; x0.W = IMGS;

  // backbone
  Act c1 = cbs_apply(R, takeCBS(cur), x0, 64, 3, 2, nullptr, 0, 0);    // stem  192
  Act t  = cbs_apply(R, takeCBS(cur), c1, 128, 3, 2, nullptr, 0, 0);   // s1c   96
  Act c2 = c2f_run(R, cur, t, 128, 3, true);                           // s1f
  t  = cbs_apply(R, takeCBS(cur), c2, 256, 3, 2, nullptr, 0, 0);       // s2c   48
  Act c3 = c2f_run(R, cur, t, 256, 6, true);                           // s2f
  t  = cbs_apply(R, takeCBS(cur), c3, 512, 3, 2, nullptr, 0, 0);       // s3c   24
  Act c4 = c2f_run(R, cur, t, 512, 6, true);                           // s3f
  t  = cbs_apply(R, takeCBS(cur), c4, 512, 3, 2, nullptr, 0, 0);       // s4c   12
  Act c5a = c2f_run(R, cur, t, 512, 3, true);                          // s4f
  Act p5  = sppf_run(R, cur, c5a, 512);                                // sppf  12

  // FPN
  Act p4, p3;
  {
    int H = 24, W = 24, Ct = 1024;
    bf16_t* cat = R.ah((size_t)NB * Ct * H * W);
    size_t tot = (size_t)NB * 512 * H * W;
    k_upsample2<<<gr1(tot), 256, 0, stream>>>(p5.p, p5.c0, p5.Ct, cat, 0, Ct,
                                              NB, 512, 12, 12);
    cbs_apply(R, takeCBS(cur), c4, 512, 1, 1, cat, 512, Ct);           // lat1
    Act catA; catA.p = cat; catA.c0 = 0; catA.Ct = Ct; catA.C = Ct; catA.H = H; catA.W = W;
    p4 = c2f_run(R, cur, catA, 512, 3, false);                         // fpn1
  }
  {
    int H = 48, W = 48, Ct = 768;
    bf16_t* cat = R.ah((size_t)NB * Ct * H * W);
    size_t tot = (size_t)NB * 512 * H * W;
    k_upsample2<<<gr1(tot), 256, 0, stream>>>(p4.p, p4.c0, p4.Ct, cat, 0, Ct,
                                              NB, 512, 24, 24);
    cbs_apply(R, takeCBS(cur), c3, 256, 1, 1, cat, 512, Ct);           // lat2
    Act catA; catA.p = cat; catA.c0 = 0; catA.Ct = Ct; catA.C = Ct; catA.H = H; catA.W = W;
    p3 = c2f_run(R, cur, catA, 256, 3, false);                         // fpn2
  }
  Act n3 = p3;

  // PAN
  Act n4, n5;
  {
    int H = 24, W = 24, Ct = 768;
    bf16_t* cat = R.ah((size_t)NB * Ct * H * W);
    cbs_apply(R, takeCBS(cur), n3, 256, 3, 2, cat, 0, Ct);             // down1
    size_t tot = (size_t)NB * 512 * H * W;
    k_copy_slice<<<gr1(tot), 256, 0, stream>>>(p4.p, p4.c0, p4.Ct, cat, 256, Ct,
                                               NB, 512, H * W);
    Act catA; catA.p = cat; catA.c0 = 0; catA.Ct = Ct; catA.C = Ct; catA.H = H; catA.W = W;
    n4 = c2f_run(R, cur, catA, 512, 3, false);                         // pan1
  }
  {
    int H = 12, W = 12, Ct = 1024;
    bf16_t* cat = R.ah((size_t)NB * Ct * H * W);
    cbs_apply(R, takeCBS(cur), n4, 512, 3, 2, cat, 0, Ct);             // down2
    size_t tot = (size_t)NB * 512 * H * W;
    k_copy_slice<<<gr1(tot), 256, 0, stream>>>(p5.p, p5.c0, p5.Ct, cat, 512, Ct,
                                               NB, 512, H * W);
    Act catA; catA.p = cat; catA.c0 = 0; catA.Ct = Ct; catA.C = Ct; catA.H = H; catA.W = W;
    n5 = c2f_run(R, cur, catA, 512, 3, false);                         // pan2
  }

  // output layout: dets(3) + refs(3) + props(3)
  float* out = (float*)d_out;
  size_t d0 = (size_t)NB * 11 * 48 * 48;
  size_t d1 = (size_t)NB * 11 * 24 * 24;
  size_t d2 = (size_t)NB * 11 * 12 * 12;
  float* det0 = out;
  float* det1 = det0 + d0;
  float* det2 = det1 + d1;
  float* ref0 = det2 + d2;
  float* ref1 = ref0 + (size_t)NB * KTOP * 10;
  float* ref2 = ref1 + (size_t)NB * KTOP * 10;
  float* prop0 = ref2 + (size_t)NB * KTOP * 10;
  float* prop1 = prop0 + (size_t)NB * KTOP * 5;
  float* prop2 = prop1 + (size_t)NB * KTOP * 5;

  run_det(R, cur, n3, 256, det0);   // det_p3
  run_det(R, cur, n4, 512, det1);   // det_p4
  run_det(R, cur, n5, 512, det2);   // det_p5

  run_post(R, cur, det0, n3, 8.f,  ref0, prop0);   // ref_p3
  run_post(R, cur, det1, n4, 16.f, ref1, prop1);   // ref_p4
  run_post(R, cur, det2, n5, 32.f, ref2, prop2);   // ref_p5
}
